// DeformMCALayer_32040456028212
// MI455X (gfx1250) — matrix operand order repaired and checked by reading, not compiled, not run on hardware
//
#include <hip/hip_runtime.h>
#include <hip/hip_bf16.h>
#include <math.h>

// ---------------------------------------------------------------------------
// Problem constants (from reference): B=8, C=256, H=W=64, Cout=256, 3x3 taps
// ---------------------------------------------------------------------------
#define BATCH   8
#define CIN     256
#define COUT    256
#define HH      64
#define WW      64
#define HW      (HH * WW)          // 4096
#define KTAPS   9

// Block tile for the main WMMA GEMM: 64(M) x 256(N), 8 waves of 32x64
#define MT      64                 // M tile (pixels)
#define NT      256                // N tile (out channels) == COUT
#define LAP     8                  // LDS row padding (bf16 elems) vs bank conflicts

typedef __attribute__((ext_vector_type(16))) __bf16 v16bf;
typedef __attribute__((ext_vector_type(4)))  __bf16 v4bf;
typedef __attribute__((ext_vector_type(8)))  float  v8f;
typedef __attribute__((ext_vector_type(4)))  float  v4f;

union AFrag { unsigned int u[8]; v16bf v; };
union BFrag { unsigned int u[8]; v16bf v; };

// ---------------------------------------------------------------------------
// Workspace layout (bytes)
// ---------------------------------------------------------------------------
#define WS_OFF_OFFSET   ((size_t)0)                         // offsets fp32 [B,18,HW]   2.36 MB
#define WS_XH_OFFSET    ((size_t)2359296)                   // x NHWC bf16 [B,H,W,C]   16.78 MB
#define WS_WT_OFFSET    ((size_t)(2359296 + 16777216))      // wt bf16 [tap,c,o]        1.18 MB
#define WS_Y_OFFSET     ((size_t)(19136512 + 1179648))      // y fp32 [B,Cout,HW]      33.55 MB
#define WS_SUM_OFFSET   ((size_t)(20316160 + 33554432))     // sums fp32 [B*Cout]
#define WS_SQ_OFFSET    (WS_SUM_OFFSET + 8192)
#define WS_ATTN_OFFSET  (WS_SQ_OFFSET + 8192)

// ---------------------------------------------------------------------------
// Kernel 0: zero the sum / sumsq accumulators (ws is poisoned by harness)
// ---------------------------------------------------------------------------
__global__ __launch_bounds__(256) void dk_init_sums(float* __restrict__ sums,
                                                    float* __restrict__ sumsq) {
    int i = blockIdx.x * blockDim.x + threadIdx.x;
    if (i < BATCH * COUT) { sums[i] = 0.0f; sumsq[i] = 0.0f; }
}

// ---------------------------------------------------------------------------
// Kernel 1: pack x NCHW fp32 -> NHWC bf16 (coalesced writes; reads hit L2)
// ---------------------------------------------------------------------------
__global__ __launch_bounds__(256) void dk_pack_x(const float* __restrict__ x,
                                                 __bf16* __restrict__ xh) {
    int i = blockIdx.x * blockDim.x + threadIdx.x;       // over B*H*W*C
    if (i >= BATCH * HW * CIN) return;
    int c    = i & (CIN - 1);
    int rest = i >> 8;                                   // b*HW + p
    int p    = rest & (HW - 1);
    int b    = rest >> 12;
    xh[i] = (__bf16)x[((b * CIN + c) * HW) + p];
}

// ---------------------------------------------------------------------------
// Kernel 2: pack deform_w [o,c,ky,kx] fp32 -> wt[tap][c][o] bf16
// ---------------------------------------------------------------------------
__global__ __launch_bounds__(256) void dk_pack_w(const float* __restrict__ w,
                                                 __bf16* __restrict__ wt) {
    int i = blockIdx.x * blockDim.x + threadIdx.x;       // over 9*256*256
    if (i >= KTAPS * CIN * COUT) return;
    int o    = i & (COUT - 1);
    int rest = i >> 8;
    int c    = rest & (CIN - 1);
    int tap  = rest >> 8;
    wt[i] = (__bf16)w[((o * CIN + c) * KTAPS) + tap];
}

// ---------------------------------------------------------------------------
// Kernel 3: offset-predicting conv (direct, fp32). One block per output row.
// ---------------------------------------------------------------------------
__global__ __launch_bounds__(64) void dk_offset_conv(const float* __restrict__ x,
                                                     const float* __restrict__ ow,
                                                     float* __restrict__ off) {
    __shared__ float w[CIN * KTAPS];                     // 2304 floats
    int blk  = blockIdx.x;                               // B*18*64 blocks
    int row  = blk & (HH - 1);
    int rest = blk >> 6;
    int ko   = rest % (2 * KTAPS);
    int b    = rest / (2 * KTAPS);

    for (int i = threadIdx.x; i < CIN * KTAPS; i += 64)
        w[i] = ow[ko * (CIN * KTAPS) + i];
    __syncthreads();

    int px = threadIdx.x;
    float acc = 0.0f;
    const float* xb = x + (size_t)b * CIN * HW;
    for (int c = 0; c < CIN; ++c) {
        const float* xc = xb + c * HW;
        const float* wc = w + c * KTAPS;
#pragma unroll
        for (int ky = 0; ky < 3; ++ky) {
            int yy = row + ky - 1;
            if (yy < 0 || yy >= HH) continue;
#pragma unroll
            for (int kx = 0; kx < 3; ++kx) {
                int xx = px + kx - 1;
                if (xx < 0 || xx >= WW) continue;
                acc = fmaf(xc[yy * WW + xx], wc[ky * 3 + kx], acc);
            }
        }
    }
    off[((b * (2 * KTAPS) + ko) * HW) + row * WW + px] = acc;
}

// ---------------------------------------------------------------------------
// Kernel 4: fused deformable gather + WMMA GEMM (bf16 in, fp32 acc)
//   grid = (M_tiles=64, 1, B=8), block = 256 (8 waves, 2Mx4N of 32x64 tiles)
//   Per K-step: B tile via global_load_async_to_lds_b128 (ASYNCcnt DMA,
//   overlapped with the bilinear gather), then 8 independent
//   v_wmma_f32_16x16x32_bf16 per wave. 72 K-steps (9 taps x 8 c-blocks).
// ---------------------------------------------------------------------------
__global__ __launch_bounds__(256) void dk_deform_wmma(
        const __bf16* __restrict__ xh,   // [B,H,W,C] bf16
        const float*  __restrict__ off,  // [B,18,HW]
        const __bf16* __restrict__ wt,   // [tap,c,o] bf16
        float* __restrict__ y,           // [B,Cout,HW]
        float* __restrict__ sums,
        float* __restrict__ sumsq) {
    __shared__ __align__(16) __bf16 lA[MT][32 + LAP];    // ~5 KB : M x K tile
    __shared__ __align__(16) __bf16 lB[32][NT + LAP];    // ~17 KB: K x N tile
    __shared__ int   sY0[MT], sX0[MT];
    __shared__ float sWt[MT][4];                         // bilinear corner weights

    const int tid  = threadIdx.x;
    const int lane = tid & 31;
    const int w    = tid >> 5;
    const int wm   = w >> 2;                             // 0..1 -> +32 M rows
    const int wn   = w & 3;                              // 0..3 -> +64 N cols

    const int b  = blockIdx.z;
    const int m0 = blockIdx.x * MT;                      // pixel base
    const int n0 = blockIdx.y * NT;                      // out-channel base (0)

    const __bf16* xb = xh + ((size_t)b << 20);           // b * H*W*C

    v8f acc[2][4];
#pragma unroll
    for (int i = 0; i < 2; ++i)
#pragma unroll
        for (int j = 0; j < 4; ++j)
            acc[i][j] = (v8f){0.f,0.f,0.f,0.f,0.f,0.f,0.f,0.f};

    for (int tap = 0; tap < KTAPS; ++tap) {
        const int ky = tap / 3, kx = tap % 3;
        // ---- per-(pixel, tap) bilinear metadata, shared by all 8 c-blocks
        if (tid < MT) {
            int p  = m0 + tid;
            int py = p >> 6, px = p & (WW - 1);
            float dy = off[((b * (2 * KTAPS) + 2 * tap + 0) * HW) + p];
            float dx = off[((b * (2 * KTAPS) + 2 * tap + 1) * HW) + p];
            float sy = (float)(py - 1 + ky) + dy;
            float sx = (float)(px - 1 + kx) + dx;
            float fy = floorf(sy), fx = floorf(sx);
            float ly = sy - fy,   lx = sx - fx;
            sY0[tid] = (int)fy;  sX0[tid] = (int)fx;
            sWt[tid][0] = (1.f - ly) * (1.f - lx);
            sWt[tid][1] = (1.f - ly) * lx;
            sWt[tid][2] = ly * (1.f - lx);
            sWt[tid][3] = ly * lx;
        }
        __syncthreads();

        for (int cb = 0; cb < 8; ++cb) {
            const int c0 = cb << 5;                      // channel base of K-block

            // ---- kick off async DMA of the B tile: 32 x 256 bf16 (16 KB),
            //      64 bytes per thread, straight into LDS (no VGPR round trip)
            {
                int r = tid >> 3;                        // 0..31 (channel row)
                int o = (tid & 7) << 5;                  // 0..224 step 32
                const __bf16* srcp =
                    &wt[(((size_t)tap * CIN + c0 + r) << 8) + n0 + o];
                unsigned           ldsa = (unsigned)(uintptr_t)&lB[r][o];
                unsigned long long ga   = (unsigned long long)(uintptr_t)srcp;
                asm volatile(
                    "global_load_async_to_lds_b128 %0, %1, off\n\t"
                    "global_load_async_to_lds_b128 %0, %1, off offset:16\n\t"
                    "global_load_async_to_lds_b128 %0, %1, off offset:32\n\t"
                    "global_load_async_to_lds_b128 %0, %1, off offset:48"
                    :: "v"(ldsa), "v"(ga) : "memory");
            }

            // ---- gather A tile while the DMA runs: 64 px x 32 ch, 8 ch/thread
            {
                int m  = tid >> 2;                       // 0..63
                int cg = (tid & 3) << 3;                 // 0,8,16,24
                int y0 = sY0[m], x0 = sX0[m];
                float w00 = sWt[m][0], w01 = sWt[m][1];
                float w10 = sWt[m][2], w11 = sWt[m][3];
                float a[8];
#pragma unroll
                for (int j = 0; j < 8; ++j) a[j] = 0.f;
                int cidx = c0 + cg;
                auto corner = [&](int yy, int xx, float wg) {
                    if (yy >= 0 && yy < HH && xx >= 0 && xx < WW) {
                        const v4bf* p = reinterpret_cast<const v4bf*>(
                            xb + ((((yy << 6) + xx) << 8) + cidx));
                        v4bf v0 = p[0], v1 = p[1];
#pragma unroll
                        for (int j = 0; j < 4; ++j) {
                            a[j]     = fmaf(wg, (float)v0[j], a[j]);
                            a[j + 4] = fmaf(wg, (float)v1[j], a[j + 4]);
                        }
                    }
                };
                corner(y0,     x0,     w00);
                corner(y0,     x0 + 1, w01);
                corner(y0 + 1, x0,     w10);
                corner(y0 + 1, x0 + 1, w11);
                v4bf p0 = { (__bf16)a[0], (__bf16)a[1], (__bf16)a[2], (__bf16)a[3] };
                v4bf p1 = { (__bf16)a[4], (__bf16)a[5], (__bf16)a[6], (__bf16)a[7] };
                *reinterpret_cast<v4bf*>(&lA[m][cg])     = p0;
                *reinterpret_cast<v4bf*>(&lA[m][cg + 4]) = p1;
            }

            // ---- drain the DMA, then block-wide barrier
            asm volatile("s_wait_asynccnt 0x0" ::: "memory");
            __syncthreads();

            // ---- build fragments per CDNA5 wave32 WMMA layouts
            AFrag af[2]; BFrag bf[4];
            {
                const int kh = (lane >> 4) << 2;         // uint offset: 0 or 4
#pragma unroll
                for (int i = 0; i < 2; ++i) {
                    const unsigned int* rowA = reinterpret_cast<const unsigned int*>(
                        &lA[(wm << 5) + (i << 4) + (lane & 15)][0]);
#pragma unroll
                    for (int v = 0; v < 4; ++v) {
                        af[i].u[v]     = rowA[kh + v];
                        af[i].u[v + 4] = rowA[8 + kh + v];
                    }
                }
                const int nB = (wn << 6) + (lane & 15);
                const int kb = (lane >> 4) << 3;
#pragma unroll
                for (int j = 0; j < 4; ++j)
#pragma unroll
                    for (int e = 0; e < 16; ++e)
                        bf[j].v[e] = lB[kb + (e & 7) + ((e >> 3) << 4)][nB + (j << 4)];
            }

            // ---- 8 independent WMMAs back-to-back (keep the XDL pipe busy)
#pragma unroll
            for (int i = 0; i < 2; ++i)
#pragma unroll
                for (int j = 0; j < 4; ++j)
                    acc[i][j] = __builtin_amdgcn_wmma_f32_16x16x32_bf16(
                        false, af[i].v, false, bf[j].v, (short)0,
                        acc[i][j], false, false);

            __syncthreads();
        }
    }

    // ---- writeback per C/D layout: lane 0-15 N=lane, M=r; lane 16-31 M=r+8
    {
        const int ocb = n0 + (wn << 6) + (lane & 15);    // j=0 out channel
        const int mbb = m0 + (wm << 5) + ((lane >> 4) << 3);
#pragma unroll
        for (int j = 0; j < 4; ++j) {
            int oc = ocb + (j << 4);
            float s = 0.f, q = 0.f;
#pragma unroll
            for (int i = 0; i < 2; ++i) {
                v8f a = acc[i][j];
                int mb = mbb + (i << 4);
                v4f lo = {a[0], a[1], a[2], a[3]};
                v4f hi = {a[4], a[5], a[6], a[7]};
                float* dst = &y[((size_t)(b * COUT + oc) * HW) + mb];
                *reinterpret_cast<v4f*>(dst)     = lo;
                *reinterpret_cast<v4f*>(dst + 4) = hi;
#pragma unroll
                for (int r = 0; r < 8; ++r) { s += a[r]; q = fmaf(a[r], a[r], q); }
            }
            atomicAdd(&sums [b * COUT + oc], s);
            atomicAdd(&sumsq[b * COUT + oc], q);
        }
    }
}

// ---------------------------------------------------------------------------
// Kernel 5: attn[b,o] = sigmoid(mean + std)  (ddof=1, N=4096)
// ---------------------------------------------------------------------------
__global__ __launch_bounds__(256) void dk_attn(const float* __restrict__ sums,
                                               const float* __restrict__ sumsq,
                                               float* __restrict__ attn) {
    int i = blockIdx.x * blockDim.x + threadIdx.x;
    if (i >= BATCH * COUT) return;
    const float n = (float)HW;
    float s  = sums[i];
    float mean = s / n;
    float var  = (sumsq[i] - s * s / n) / (n - 1.0f);
    float sd   = sqrtf(fmaxf(var, 0.0f));
    float a    = mean + sd;
    attn[i] = 1.0f / (1.0f + expf(-a));
}

// ---------------------------------------------------------------------------
// Kernel 6: out = y * attn (broadcast over spatial)
// ---------------------------------------------------------------------------
__global__ __launch_bounds__(256) void dk_scale(const float* __restrict__ y,
                                                const float* __restrict__ attn,
                                                float* __restrict__ out) {
    int i = blockIdx.x * blockDim.x + threadIdx.x;
    if (i >= BATCH * COUT * HW) return;
    out[i] = y[i] * attn[i >> 12];                       // i/4096 -> (b*Cout+o)
}

// ---------------------------------------------------------------------------
// Launcher
// ---------------------------------------------------------------------------
extern "C" void kernel_launch(void* const* d_in, const int* in_sizes, int n_in,
                              void* d_out, int out_size, void* d_ws, size_t ws_size,
                              hipStream_t stream) {
    const float* x        = (const float*)d_in[0];
    const float* offset_w = (const float*)d_in[1];
    const float* deform_w = (const float*)d_in[2];
    float* out = (float*)d_out;

    char* ws      = (char*)d_ws;
    float*  off_f = (float*)(ws + WS_OFF_OFFSET);
    __bf16* xh    = (__bf16*)(ws + WS_XH_OFFSET);
    __bf16* wt    = (__bf16*)(ws + WS_WT_OFFSET);
    float*  yws   = (float*)(ws + WS_Y_OFFSET);
    float*  sums  = (float*)(ws + WS_SUM_OFFSET);
    float*  sumsq = (float*)(ws + WS_SQ_OFFSET);
    float*  attn  = (float*)(ws + WS_ATTN_OFFSET);

    dk_init_sums<<<(BATCH * COUT + 255) / 256, 256, 0, stream>>>(sums, sumsq);
    dk_pack_x<<<(BATCH * HW * CIN + 255) / 256, 256, 0, stream>>>(x, xh);
    dk_pack_w<<<(KTAPS * CIN * COUT + 255) / 256, 256, 0, stream>>>(deform_w, wt);
    dk_offset_conv<<<BATCH * 2 * KTAPS * HH, 64, 0, stream>>>(x, offset_w, off_f);

    dim3 grid(HW / MT, COUT / NT, BATCH);                // 64 x 1 x 8
    dk_deform_wmma<<<grid, 256, 0, stream>>>(xh, off_f, wt, yws, sums, sumsq);

    dk_attn<<<(BATCH * COUT + 255) / 256, 256, 0, stream>>>(sums, sumsq, attn);
    dk_scale<<<(BATCH * COUT * HW + 255) / 256, 256, 0, stream>>>(yws, attn, out);
}
